// UGNN_74947179315406
// MI455X (gfx1250) — matrix operand order repaired and checked by reading, not compiled, not run on hardware
//
#include <hip/hip_runtime.h>

typedef __bf16 bf16_t;
typedef __attribute__((ext_vector_type(16))) __bf16 v16bf;
typedef __attribute__((ext_vector_type(8)))  float  v8f;

// ---------------------------------------------------------------------------
// CDNA5 async global->LDS copy (ASYNCcnt-tracked) + wait, via inline asm.
// Per-lane: copies 16 bytes from the lane's global address to the lane's LDS
// address. Generic shared-pointer low 32 bits are the LDS byte offset (ISA
// aperture rule: low 32 address bits give the LDS address).
// ---------------------------------------------------------------------------
__device__ __forceinline__ void async_copy_b128(unsigned lds_off, const void* gaddr) {
  asm volatile("global_load_async_to_lds_b128 %0, %1, off"
               :: "v"(lds_off), "v"(gaddr) : "memory");
}
__device__ __forceinline__ void wait_async0() {
  asm volatile("s_wait_asynccnt 0" ::: "memory");
}

// ---------------------------------------------------------------------------
// Generic bf16 WMMA GEMM: C[M,N] (f32) = A[M,K] (bf16,row-major) * B[K,N]
// 256 threads = 8 waves; block tile 128x64; K-step 32; batched via blockIdx.z.
// Double-buffered LDS: A tile staged by async DMA, B tile staged transposed.
// Requirements (all call sites satisfy): K % 32 == 0, N even, and B buffer
// readable up to row K-1, column n0+63 (slack or exact fit with N >= 64).
// Garbage in A rows >= M / B cols >= N only reaches discarded C elements.
// ---------------------------------------------------------------------------
#define GBM 128
#define GBN 64
#define GBK 32
#define GPAD 8

__global__ __launch_bounds__(256) void k_gemm_bf16(
    const bf16_t* __restrict__ Ag, const bf16_t* __restrict__ Bg,
    float* __restrict__ Cg, int M, int N, int K, long sA, long sB, long sC) {
  __shared__ __align__(16) bf16_t As[2][GBM][GBK + GPAD];
  __shared__ __align__(16) bf16_t Bs[2][GBN][GBK + GPAD];   // transposed: [n][k]
  const int tid  = threadIdx.x;
  const int lane = tid & 31;
  const int wv   = tid >> 5;
  const bf16_t* A = Ag + (long)blockIdx.z * sA;
  const bf16_t* B = Bg + (long)blockIdx.z * sB;
  float*        C = Cg + (long)blockIdx.z * sC;
  const int m0 = blockIdx.y * GBM;
  const int n0 = blockIdx.x * GBN;

  v8f acc[4] = {};

  auto load_tiles = [&](int buf, int k0) {
    // A tile 128x32 bf16 = 512 x 16B chunks, 2 per thread, async DMA to LDS.
#pragma unroll
    for (int i = 0; i < 2; ++i) {
      int e  = tid + i * 256;
      int r  = e >> 2;               // row in tile
      int ch = (e & 3) << 3;         // bf16 offset of this 16B chunk
      int gm = m0 + r;
      if (gm < M) {
        unsigned lds = (unsigned)(size_t)&As[buf][r][ch];
        async_copy_b128(lds, (const void*)(A + (long)gm * K + (k0 + ch)));
      }
    }
    // B tile 32x64: unconditional dword loads, stored transposed as Bs[n][k].
#pragma unroll
    for (int i = 0; i < 4; ++i) {
      int e  = tid + i * 256;
      int kk = e >> 5;
      int n  = (e & 31) << 1;
      union { unsigned u; bf16_t h[2]; } cv;
      cv.u = *(const unsigned*)(B + (long)(k0 + kk) * N + (n0 + n));
      Bs[buf][n][kk]     = cv.h[0];
      Bs[buf][n + 1][kk] = cv.h[1];
    }
    // pull the B tile two K-steps ahead into L2 (global_prefetch_b8)
    int kpf = k0 + 2 * GBK;
    if (kpf < K)
      __builtin_prefetch(B + (long)(kpf + (tid >> 5)) * N + n0 + 2 * (tid & 31), 0, 1);
  };

  load_tiles(0, 0);
  int cur = 0;
  for (int k0 = 0; k0 < K; k0 += GBK) {
    wait_async0();        // our async DMAs into As[cur] are done
    __syncthreads();      // everyone's stores/DMAs into buffer `cur` visible
    int nk = k0 + GBK;
    if (nk < K) load_tiles(cur ^ 1, nk);   // overlap next tile with compute

    // A fragment (16x32, ISA 7.12.2 layout): two ds_load_b128 per wave-tile
    union AU { uint4 q[2]; v16bf v; } a;
    const int mrow  = (wv << 4) + (lane & 15);
    const int khalf = (lane >> 4) << 3;                  // 0 or 8
    a.q[0] = *(const uint4*)&As[cur][mrow][khalf];       // K khalf..khalf+7
    a.q[1] = *(const uint4*)&As[cur][mrow][16 + khalf];  // K 16+khalf..
#pragma unroll
    for (int f = 0; f < 4; ++f) {
      union BU { uint4 q[2]; v16bf v; } b;
      const int ncol = (f << 4) + (lane & 15);
      const int kb   = (lane >> 4) << 3;                 // 0 or 8
      b.q[0] = *(const uint4*)&Bs[cur][ncol][kb];
      b.q[1] = *(const uint4*)&Bs[cur][ncol][16 + kb];
      acc[f] = __builtin_amdgcn_wmma_f32_16x16x32_bf16(
          false, a.v, false, b.v, (short)0, acc[f], false, false);
    }
    cur ^= 1;
  }

  // C/D layout: VGPR i -> row i (lanes 0-15) / row i+8 (lanes 16-31), col lane%16
  const int rbase = m0 + (wv << 4) + ((lane >> 4) << 3);
  const int cl    = lane & 15;
#pragma unroll
  for (int f = 0; f < 4; ++f) {
    int gn = n0 + (f << 4) + cl;
    if (gn < N) {
#pragma unroll
      for (int i = 0; i < 8; ++i) {
        int gm = rbase + i;
        if (gm < M) C[(long)gm * N + gn] = acc[f][i];
      }
    }
  }
}

// ---------------------------------------------------------------------------
// Elementwise / transform kernels
// ---------------------------------------------------------------------------
__global__ void k_f32_to_bf16(const float* __restrict__ s, bf16_t* __restrict__ d, long n) {
  long i = (long)blockIdx.x * 256 + threadIdx.x;
  if (i < n) d[i] = (bf16_t)s[i];
}

__global__ void k_wt_enc(const float* __restrict__ w, bf16_t* __restrict__ o, long n) {
  long i = (long)blockIdx.x * 256 + threadIdx.x;   // w:[512,1,64] -> o:[64,512]
  if (i >= n) return;
  int co = (int)(i / 64), k = (int)(i % 64);
  o[(long)k * 512 + co] = (bf16_t)w[i];
}

__global__ void k_wt_conv(const float* __restrict__ w, bf16_t* __restrict__ o,
                          int cin, int cout, long n) {
  long i = (long)blockIdx.x * 256 + threadIdx.x;   // w:[cout,cin,3] -> o:[3*cin,cout]
  if (i >= n) return;
  int co = (int)(i / ((long)cin * 3));
  int r  = (int)(i % ((long)cin * 3));
  int ci = r / 3, k = r % 3;
  o[((long)k * cin + ci) * cout + co] = (bf16_t)w[i];
}

__global__ void k_wt_up(const float* __restrict__ w, bf16_t* __restrict__ o,
                        int cin, int cout, long n) {
  long i = (long)blockIdx.x * 256 + threadIdx.x;   // w:[cin,cout,2] -> o:[cin,2*cout]
  if (i >= n) return;
  int ci = (int)(i / ((long)cout * 2));
  int r  = (int)(i % ((long)cout * 2));
  int co = r / 2, j = r % 2;
  o[(long)ci * (2 * cout) + j * cout + co] = (bf16_t)w[i];
}

__global__ void k_enc_im2col(const float* __restrict__ x, bf16_t* __restrict__ A,
                             int Bn, int Lout, int Lx, long n) {
  long i = (long)blockIdx.x * 256 + threadIdx.x;
  if (i >= n) return;
  int k = (int)(i % 64);
  long bt = i / 64;
  int t = (int)(bt % Lout), b = (int)(bt / Lout);
  A[i] = (bf16_t)x[(long)b * Lx + (long)t * 32 + k];
}

__global__ void k_im2col3(const bf16_t* __restrict__ act, bf16_t* __restrict__ o,
                          int Bn, int L, int C, long n) {
  long i = (long)blockIdx.x * 256 + threadIdx.x;
  if (i >= n) return;
  int c = (int)(i % C);
  long r = i / C;
  int k = (int)(r % 3);
  long bt = r / 3;
  int t = (int)(bt % L), b = (int)(bt / L);
  int ts = t - 1 + k;
  bf16_t v = (bf16_t)0.f;
  if (ts >= 0 && ts < L) v = act[((long)b * L + ts) * C + c];
  o[i] = v;
}

__global__ void k_bn_stats(const float* __restrict__ X, float* __restrict__ mean,
                           float* __restrict__ rstd, long rows, int C) {
  __shared__ float s1[256], s2[256];
  int c = blockIdx.x;
  float a = 0.f, b = 0.f;
  for (long r = threadIdx.x; r < rows; r += 256) {
    float v = X[r * C + c];
    a += v; b += v * v;
  }
  s1[threadIdx.x] = a; s2[threadIdx.x] = b;
  __syncthreads();
  for (int st = 128; st > 0; st >>= 1) {
    if ((int)threadIdx.x < st) { s1[threadIdx.x] += s1[threadIdx.x + st];
                                 s2[threadIdx.x] += s2[threadIdx.x + st]; }
    __syncthreads();
  }
  if (threadIdx.x == 0) {
    float mu  = s1[0] / (float)rows;
    float var = s2[0] / (float)rows - mu * mu;
    mean[c] = mu;
    rstd[c] = rsqrtf(var + 1e-5f);
  }
}

__global__ void k_bn_apply_relu(const float* __restrict__ X, const float* __restrict__ mean,
                                const float* __restrict__ rstd, const float* __restrict__ g,
                                const float* __restrict__ bt, float* __restrict__ Yf,
                                bf16_t* __restrict__ Yh, long n, int C) {
  long i = (long)blockIdx.x * 256 + threadIdx.x;
  if (i >= n) return;
  int c = (int)(i % C);
  float v = (X[i] - mean[c]) * rstd[c] * g[c] + bt[c];
  v = fmaxf(v, 0.f);
  if (Yf) Yf[i] = v;
  Yh[i] = (bf16_t)v;
}

__global__ void k_maxpool(const bf16_t* __restrict__ in, bf16_t* __restrict__ out,
                          int Bn, int Lin, int C, long n) {
  long i = (long)blockIdx.x * 256 + threadIdx.x;
  if (i >= n) return;
  int Lout = Lin >> 1;
  int c = (int)(i % C);
  long bt = i / C;
  int t = (int)(bt % Lout), b = (int)(bt / Lout);
  const bf16_t* p = in + ((long)b * Lin + 2 * t) * C + c;
  out[i] = (bf16_t)fmaxf((float)p[0], (float)p[C]);
}

__global__ void k_concat_pad(const bf16_t* __restrict__ skip, const bf16_t* __restrict__ up,
                             bf16_t* __restrict__ out, int Bn, int Lout, int Lup,
                             int Cs, int Cu, long n) {
  long i = (long)blockIdx.x * 256 + threadIdx.x;
  if (i >= n) return;
  int Ct = Cs + Cu;
  int c = (int)(i % Ct);
  long bt = i / Ct;
  int t = (int)(bt % Lout), b = (int)(bt / Lout);
  bf16_t v;
  if (c < Cs) v = skip[((long)b * Lout + t) * Cs + c];
  else        v = (t < Lup) ? up[((long)b * Lup + t) * Cu + (c - Cs)] : (bf16_t)0.f;
  out[i] = v;
}

// transpose [R,Cc] -> [Cc,Rp] per batch (Rp >= R; pad columns left untouched)
__global__ void k_transpose_pad_bf16(const bf16_t* __restrict__ in, bf16_t* __restrict__ out,
                                     int R, int Cc, int Rp) {
  long perIn  = (long)R * Cc;
  long perOut = (long)Cc * Rp;
  const bf16_t* ip = in + (long)blockIdx.z * perIn;
  bf16_t*       op = out + (long)blockIdx.z * perOut;
  long i = (long)blockIdx.x * 256 + threadIdx.x;
  if (i >= perIn) return;
  int c = (int)(i % Cc);
  long r = i / Cc;
  op[(long)c * Rp + r] = ip[i];
}

__global__ void k_sqnorm(const float* __restrict__ x, float* __restrict__ sq, int C) {
  __shared__ float s[256];
  long node = blockIdx.x;
  float a = 0.f;
  for (int c = threadIdx.x; c < C; c += 256) { float v = x[node * C + c]; a += v * v; }
  s[threadIdx.x] = a;
  __syncthreads();
  for (int st = 128; st > 0; st >>= 1) {
    if ((int)threadIdx.x < st) s[threadIdx.x] += s[threadIdx.x + st];
    __syncthreads();
  }
  if (threadIdx.x == 0) sq[node] = s[0];
}

__global__ void k_topk(const float* __restrict__ G, const float* __restrict__ sq,
                       int* __restrict__ idx, int Bn, int L, int ldg, long n) {
  long i = (long)blockIdx.x * 256 + threadIdx.x;
  if (i >= n) return;
  int b = (int)(i / L), r = (int)(i % L);
  const float* g  = G + ((long)b * L + r) * ldg;
  const float* sb = sq + (long)b * L;
  float si = sb[r];
  int chosen[8];
  for (int kk = 0; kk < 8; ++kk) {
    float best = 3.4e38f; int bj = 0;
    for (int j = 0; j < L; ++j) {
      if (j == r) continue;
      bool used = false;
      for (int q = 0; q < kk; ++q) if (chosen[q] == j) used = true;
      if (used) continue;
      float d = si + sb[j] - 2.f * g[j];
      if (d < best) { best = d; bj = j; }
    }
    chosen[kk] = bj;
    idx[i * 8 + kk] = bj;
  }
}

__global__ void k_deg_init(float* __restrict__ deg, long n) {
  long i = (long)blockIdx.x * 256 + threadIdx.x;
  if (i < n) deg[i] = 9.f;   // 8 (as knn target) + 1 (self loop)
}

__global__ void k_deg_acc(const int* __restrict__ idx, float* __restrict__ deg,
                          int Bn, int L, long n) {
  long e = (long)blockIdx.x * 256 + threadIdx.x;
  if (e >= n) return;
  int b = (int)(e / ((long)L * 8));
  atomicAdd(&deg[(long)b * L + idx[e]], 1.f);
}

__global__ void k_dinv(const float* __restrict__ deg, float* __restrict__ dinv, long n) {
  long i = (long)blockIdx.x * 256 + threadIdx.x;
  if (i < n) dinv[i] = rsqrtf(deg[i]);
}

__global__ void k_gcn_init(const float* __restrict__ h, const float* __restrict__ deg,
                           const float* __restrict__ bias, float* __restrict__ out,
                           long n, int C) {
  long i = (long)blockIdx.x * 256 + threadIdx.x;
  if (i >= n) return;
  long nd = i / C;
  int c = (int)(i % C);
  out[i] = h[i] / deg[nd] + bias[c];
}

__global__ void k_gcn_edges(const float* __restrict__ h, const float* __restrict__ dinv,
                            const int* __restrict__ idx, float* __restrict__ out,
                            int Bn, int L, int C, long n) {
  long i = (long)blockIdx.x * 256 + threadIdx.x;
  if (i >= n) return;
  int c = (int)(i % C);
  long e = i / C;
  int kk = (int)(e % 8);
  long bn = e / 8;
  int r = (int)(bn % L), b = (int)(bn / L);
  int j = idx[bn * 8 + kk];
  long ni = (long)b * L + r;   // target node
  long nj = (long)b * L + j;   // neighbor node
  float w = dinv[ni] * dinv[nj];
  atomicAdd(&out[nj * C + c], h[ni * C + c] * w);   // edge tgt->nbr
  atomicAdd(&out[ni * C + c], h[nj * C + c] * w);   // edge nbr->tgt
}

__global__ void k_relu_bf16(float* __restrict__ x, bf16_t* __restrict__ xh, long n, int doRelu) {
  long i = (long)blockIdx.x * 256 + threadIdx.x;
  if (i >= n) return;
  float v = x[i];
  if (doRelu) v = fmaxf(v, 0.f);
  x[i] = v;
  xh[i] = (bf16_t)v;
}

__global__ void k_mask(const float* __restrict__ y, const float* __restrict__ ow,
                       const float* __restrict__ ob, const float* __restrict__ xenc,
                       bf16_t* __restrict__ masked, long rows) {
  __shared__ float s[64];
  __shared__ float mval;
  long row = blockIdx.x;
  int t = threadIdx.x;
  if (t < 64) s[t] = y[row * 64 + t] * ow[t];
  __syncthreads();
  if (t == 0) {
    float a = 0.f;
    for (int i = 0; i < 64; ++i) a += s[i];
    mval = 1.f / (1.f + expf(-(a + ob[0])));
  }
  __syncthreads();
  for (int c = t; c < 512; c += 256)
    masked[row * 512 + c] = (bf16_t)(xenc[row * 512 + c] * mval);
}

__global__ void k_overlap_add(const float* __restrict__ Y, float* __restrict__ out,
                              int Bn, int L, int Lx, long n) {
  long i = (long)blockIdx.x * 256 + threadIdx.x;
  if (i >= n) return;
  int s = (int)(i % Lx), b = (int)(i / Lx);
  int t0 = s >> 5, k0 = s & 31;
  const float* Yb = Y + (long)b * L * 64;
  float r = 0.f;
  if (t0 < L)  r += Yb[(long)t0 * 64 + k0];
  if (t0 >= 1) r += Yb[(long)(t0 - 1) * 64 + k0 + 32];
  out[i] = r;
}

// ---------------------------------------------------------------------------
// Host orchestration
// ---------------------------------------------------------------------------
#define L1D(kern, n, ...) \
  kern<<<dim3((unsigned)(((long)(n) + 255) / 256)), dim3(256), 0, stream>>>(__VA_ARGS__)

extern "C" void kernel_launch(void* const* d_in, const int* in_sizes, int n_in,
                              void* d_out, int out_size, void* d_ws, size_t ws_size,
                              hipStream_t stream) {
  (void)in_sizes; (void)n_in; (void)out_size; (void)ws_size;
  const int Bb = 8, LX = 131072;
  const int L1 = 4095, L2 = 2047, L3 = 1023, L4 = 511;
  const long BL1 = (long)Bb * L1, BL2 = (long)Bb * L2, BL3 = (long)Bb * L3, BL4 = (long)Bb * L4;

  auto in = [&](int i) { return (const float*)d_in[i]; };
  // input index map (setup_inputs insertion order):
  // 0:x 1:enc_w 2..9:inc 10..17:d1 18..25:d2 26..33:d3
  // 34:gw1 35:gb1 36:gw2 37:gb2 38:gw3 39:gb3
  // 40:u1_uw 41:u1_ub 42..49:u1_dc 50..59:u2 60..69:u3 70:out_w 71:out_b 72:dec_w

  char* wp = (char*)d_ws;
  auto alloc = [&](size_t bytes) -> void* {
    void* p = (void*)wp;
    wp += (bytes + 255) & ~(size_t)255;
    return p;
  };
  auto af = [&](long n) { return (float*)alloc((size_t)n * 4); };
  auto ah = [&](long n) { return (bf16_t*)alloc((size_t)n * 2); };
  auto ai = [&](long n) { return (int*)alloc((size_t)n * 4); };

  // shared scratch
  bf16_t* colbuf = ah(BL1 * 1536);      // largest im2col (inc conv1)
  bf16_t* wtbuf  = ah(524288);          // largest transformed weight (+slack for B over-read)
  float*  cvbuf  = af(2202624);         // pre-BN GEMM outputs / GCN h
  float*  meanb  = af(512);
  float*  rstdb  = af(512);
  bf16_t* midH   = ah(2202624);
  bf16_t* upH    = ah(2202624);
  bf16_t* catH   = ah(4202496);

  auto gemm = [&](const bf16_t* A, const bf16_t* Bm, float* Cm, long M, int N, int K,
                  int batch, long sA, long sB, long sC) {
    dim3 g((unsigned)((N + GBN - 1) / GBN), (unsigned)((M + GBM - 1) / GBM), (unsigned)batch);
    k_gemm_bf16<<<g, 256, 0, stream>>>(A, Bm, Cm, (int)M, N, K, sA, sB, sC);
  };

  auto conv_bn = [&](const bf16_t* act, int L, int Cin, int Cout, const float* w,
                     const float* g, const float* bt, float* outF, bf16_t* outH) {
    long rows = (long)Bb * L;
    L1D(k_wt_conv, (long)Cout * Cin * 3, w, wtbuf, Cin, Cout, (long)Cout * Cin * 3);
    L1D(k_im2col3, rows * 3 * Cin, act, colbuf, Bb, L, Cin, rows * 3 * Cin);
    gemm(colbuf, wtbuf, cvbuf, rows, Cout, 3 * Cin, 1, 0, 0, 0);
    k_bn_stats<<<dim3((unsigned)Cout), dim3(256), 0, stream>>>(cvbuf, meanb, rstdb, rows, Cout);
    L1D(k_bn_apply_relu, rows * Cout, cvbuf, meanb, rstdb, g, bt, outF, outH, rows * Cout, Cout);
  };

  auto dconv = [&](const bf16_t* act, int L, int Cin, int Cout, int q,
                   float* outF, bf16_t* outH) {
    conv_bn(act, L, Cin, Cout, in(q + 0), in(q + 2), in(q + 3), nullptr, midH);
    conv_bn(midH, L, Cout, Cout, in(q + 4), in(q + 6), in(q + 7), outF, outH);
  };

  // ---- encoder: x_enc = conv(x, enc_w, stride 32), no bias/BN ----
  float*  xencF = af(BL1 * 512);
  bf16_t* xencH = ah(BL1 * 512);
  L1D(k_wt_enc, 32768, in(1), wtbuf, 32768);
  L1D(k_enc_im2col, BL1 * 64, in(0), colbuf, Bb, L1, LX, BL1 * 64);
  gemm(colbuf, wtbuf, xencF, BL1, 512, 64, 1, 0, 0, 0);
  L1D(k_f32_to_bf16, BL1 * 512, xencF, xencH, BL1 * 512);

  // ---- contracting path ----
  bf16_t* x1H = ah(BL1 * 64);
  dconv(xencH, L1, 512, 64, 2, nullptr, x1H);

  bf16_t* p1H = ah(BL2 * 64);
  L1D(k_maxpool, BL2 * 64, x1H, p1H, Bb, L1, 64, BL2 * 64);
  bf16_t* x2H = ah(BL2 * 128);
  dconv(p1H, L2, 64, 128, 10, nullptr, x2H);

  bf16_t* p2H = ah(BL3 * 128);
  L1D(k_maxpool, BL3 * 128, x2H, p2H, Bb, L2, 128, BL3 * 128);
  bf16_t* x3H = ah(BL3 * 256);
  dconv(p2H, L3, 128, 256, 18, nullptr, x3H);

  bf16_t* p3H = ah(BL4 * 256);
  L1D(k_maxpool, BL4 * 256, x3H, p3H, Bb, L3, 256, BL4 * 256);
  float*  x4F = af(BL4 * 512);
  bf16_t* x4H = ah(BL4 * 512);
  dconv(p3H, L4, 256, 512, 26, x4F, x4H);

  // ---- kNN graph: Gram via batched WMMA GEMM (N padded 511->512) ----
  bf16_t* x4T = ah((long)Bb * 512 * 512);     // [b][512 feat][512 nodes(padded)]
  {
    long per = (long)L4 * 512;
    dim3 g((unsigned)((per + 255) / 256), 1, (unsigned)Bb);
    k_transpose_pad_bf16<<<g, 256, 0, stream>>>(x4H, x4T, L4, 512, 512);
  }
  float* G = af((long)Bb * L4 * 512);         // row stride 512, cols >=511 unused
  gemm(x4H, x4T, G, L4, 512, 512, Bb, (long)L4 * 512, (long)512 * 512, (long)L4 * 512);

  float* sq = af(BL4);
  k_sqnorm<<<dim3((unsigned)BL4), dim3(256), 0, stream>>>(x4F, sq, 512);
  int* idx = ai(BL4 * 8);
  L1D(k_topk, BL4, G, sq, idx, Bb, L4, 512, BL4);
  float* deg  = af(BL4);
  float* dinv = af(BL4);
  L1D(k_deg_init, BL4, deg, BL4);
  L1D(k_deg_acc, BL4 * 8, idx, deg, Bb, L4, BL4 * 8);
  L1D(k_dinv, BL4, deg, dinv, BL4);

  // ---- GCN layers ----
  float*  gout = af(BL4 * 512);
  bf16_t* h1H  = ah(BL4 * 32);
  bf16_t* h2H  = ah(BL4 * 32);
  bf16_t* x4pH = ah(BL4 * 512);
  auto gcn = [&](const bf16_t* xin, int Cin, int Cout, const float* w, const float* bias,
                 bf16_t* outH, int relu) {
    L1D(k_f32_to_bf16, (long)Cin * Cout, w, wtbuf, (long)Cin * Cout);
    gemm(xin, wtbuf, cvbuf, BL4, Cout, Cin, 1, 0, 0, 0);
    L1D(k_gcn_init, BL4 * Cout, cvbuf, deg, bias, gout, BL4 * Cout, Cout);
    L1D(k_gcn_edges, BL4 * 8 * Cout, cvbuf, dinv, idx, gout, Bb, L4, Cout, BL4 * 8 * Cout);
    L1D(k_relu_bf16, BL4 * Cout, gout, outH, BL4 * Cout, relu);
  };
  gcn(x4H, 512, 32, in(34), in(35), h1H, 1);
  gcn(h1H, 32, 32, in(36), in(37), h2H, 1);
  gcn(h2H, 32, 512, in(38), in(39), x4pH, 0);

  // ---- expanding path (ConvTranspose k=2 s=2 == GEMM with [Cin, 2*Cout] weight) ----
  auto up_block = [&](const bf16_t* xlow, int Llow, int Cin, const bf16_t* skip, int Lskip,
                      int q, float* outF, bf16_t* outH) {
    int Cout = Cin / 2;
    long rowsLow = (long)Bb * Llow;
    L1D(k_wt_up, (long)Cin * Cout * 2, in(q), wtbuf, Cin, Cout, (long)Cin * Cout * 2);
    gemm(xlow, wtbuf, cvbuf, rowsLow, 2 * Cout, Cin, 1, 0, 0, 0);  // [B,Llow,2,Cout]=[B,2*Llow,Cout]
    L1D(k_f32_to_bf16, rowsLow * 2 * Cout, cvbuf, upH, rowsLow * 2 * Cout);
    int Lup = 2 * Llow;
    L1D(k_concat_pad, (long)Bb * Lskip * Cin, skip, upH, catH, Bb, Lskip, Lup,
        Cout, Cout, (long)Bb * Lskip * Cin);
    dconv(catH, Lskip, Cin, Cout, q + 2, outF, outH);
  };

  bf16_t* u1H = ah(BL3 * 256);
  up_block(x4pH, L4, 512, x3H, L3, 40, nullptr, u1H);
  bf16_t* u2H = ah(BL2 * 128);
  up_block(u1H, L3, 256, x2H, L2, 50, nullptr, u2H);
  float*  yF  = af(BL1 * 64);
  bf16_t* u3H = ah(BL1 * 64);
  up_block(u2H, L2, 128, x1H, L1, 60, yF, u3H);

  // ---- sigmoid mask + masked encoder + decoder transpose-conv ----
  bf16_t* maskedH = ah(BL1 * 512);
  k_mask<<<dim3((unsigned)BL1), dim3(256), 0, stream>>>(yF, in(70), in(71), xencF, maskedH, BL1);
  L1D(k_f32_to_bf16, 512 * 64, in(72), wtbuf, 512 * 64);
  float* Ydec = af(BL1 * 64);
  gemm(maskedH, wtbuf, Ydec, BL1, 64, 512, 1, 0, 0, 0);
  L1D(k_overlap_add, (long)Bb * LX, Ydec, (float*)d_out, Bb, L1, LX, (long)Bb * LX);
}
